// GRUScratch_18906446037014
// MI455X (gfx1250) — matrix operand order repaired and checked by reading, not compiled, not run on hardware
//
#include <hip/hip_runtime.h>

// ---------------- types / geometry ----------------
typedef __attribute__((ext_vector_type(16))) __bf16 v16bf;
typedef __attribute__((ext_vector_type(8)))  __bf16 v8bf;
typedef __attribute__((ext_vector_type(8)))  float  v8f;

constexpr int Tn = 1024, Bn = 64, Dn = 512, Hn = 512;
constexpr size_t WMAT = (size_t)Dn * Hn;          // 262144 elems per weight matrix
constexpr size_t TBH  = (size_t)Tn * Bn * Hn;     // per-gate X elements

// workspace layout (bytes)
constexpr size_t BAR_OFF = 0;                               // barrier counter (256B reserved)
constexpr size_t W_OFF   = 256;                             // packed bf16 weights (6 matrices)
constexpr size_t X_OFF   = W_OFF  + 6 * WMAT * 2;           // Xr|Xz|Xh bf16
constexpr size_t H_OFF   = X_OFF  + 3 * TBH * 2;            // H     f32  [64,512]
constexpr size_t R_OFF   = H_OFF  + (size_t)Bn * Hn * 4;    // R     f32
constexpr size_t ZH_OFF  = R_OFF  + (size_t)Bn * Hn * 4;    // Z*H   f32
constexpr size_t HR_OFF  = ZH_OFF + (size_t)Bn * Hn * 4;    // H*R   bf16
constexpr size_t HBF_OFF = HR_OFF + (size_t)Bn * Hn * 2;    // H     bf16 (GEMM A copy)

constexpr int NWG = 64;      // persistent workgroups for the recurrence (128 thr each)

// ---------------- weight packer ----------------
// Pack W (row-major [K=512][N=512] f32) into WMMA-B fragments (bf16):
// fragment (tile_n, tile_k): 32 lanes x 16 bf16, lane holds N = tile_n*16 + lane%16,
// element i -> K = tile_k*32 + (i/8)*16 + (lane/16)*8 + i%8.  32B contiguous per lane.
__global__ __launch_bounds__(256) void pack_w_kernel(
    const float* w0, const float* w1, const float* w2,
    const float* w3, const float* w4, const float* w5, __bf16* dst) {
  const float* srcs[6] = {w0, w1, w2, w3, w4, w5};
  size_t tid = (size_t)blockIdx.x * 256 + threadIdx.x;
  if (tid >= 6 * WMAT) return;
  int w    = (int)(tid / WMAT);
  int r    = (int)(tid % WMAT);
  int frag = r >> 9;          // (tile_n*16 + tile_k)
  int lane = (r >> 4) & 31;
  int i    = r & 15;
  int tn   = frag >> 4;
  int tk   = frag & 15;
  int n    = tn * 16 + (lane & 15);
  int k    = tk * 32 + (i >> 3) * 16 + (lane >> 4) * 8 + (i & 7);
  dst[tid] = (__bf16)srcs[w][(size_t)k * Hn + n];
}

// ---------------- phase 1: X_g = inputs @ W_xg + b_g for ALL 3 gates ----------------
// One wave32 owns a 16(M) x 64(N) strip for all 3 gates: A fragment is converted once
// per K-tile and reused by 12 WMMAs (3 gates x 4 N-tiles). inputs read once from HBM.
__global__ __launch_bounds__(256) void xgemm_kernel(
    const float* __restrict__ inp, const __bf16* __restrict__ wpk,
    const float* __restrict__ br, const float* __restrict__ bz,
    const float* __restrict__ bh, __bf16* __restrict__ xbuf) {
  int wid  = threadIdx.x >> 5;
  int lane = threadIdx.x & 31;
  int hi   = lane >> 4;
  int lm   = lane & 15;

  int gw = blockIdx.x * 8 + wid;             // 0 .. 32767
  int mt = gw >> 3;                          // 0..4095  (16-row tile)
  int ng = gw & 7;                           // 0..7     (64-col group)

  const float* arow = inp + ((size_t)mt * 16 + lm) * Dn;

  v8f acc[3][4];
  #pragma unroll
  for (int g = 0; g < 3; ++g)
    #pragma unroll
    for (int j = 0; j < 4; ++j)
      #pragma unroll
      for (int r = 0; r < 8; ++r) acc[g][j][r] = 0.0f;

  for (int kt = 0; kt < 16; ++kt) {
    int kb = kt * 32;
    v16bf a;
    #pragma unroll
    for (int i = 0; i < 8; ++i) {
      a[i]     = (__bf16)arow[kb + hi * 8 + i];
      a[8 + i] = (__bf16)arow[kb + 16 + hi * 8 + i];
    }
    if (kt < 15)  // stream next K-tile of weights through L2 (global_prefetch_b8)
      __builtin_prefetch(wpk + (size_t)((ng * 4) * 16 + kt + 1) * 512 + lane * 16, 0, 1);
    #pragma unroll
    for (int g = 0; g < 3; ++g) {
      const __bf16* wg = wpk + (size_t)g * WMAT;
      #pragma unroll
      for (int j = 0; j < 4; ++j) {
        int tn = ng * 4 + j;
        const v16bf* bp = (const v16bf*)(wg + (size_t)(tn * 16 + kt) * 512);
        v16bf b = bp[lane];
        acc[g][j] = __builtin_amdgcn_wmma_f32_16x16x32_bf16(
            false, a, false, b, (short)0, acc[g][j], false, false);
      }
    }
  }

  const float* biases[3] = {br, bz, bh};
  #pragma unroll
  for (int g = 0; g < 3; ++g) {
    __bf16* xout = xbuf + (size_t)g * TBH;
    #pragma unroll
    for (int j = 0; j < 4; ++j) {
      int n = (ng * 4 + j) * 16 + lm;
      float bv = biases[g][n];
      #pragma unroll
      for (int r = 0; r < 8; ++r) {
        size_t mm = (size_t)mt * 16 + r + 8 * hi;
        xout[mm * Hn + n] = (__bf16)(acc[g][j][r] + bv);
      }
    }
  }
}

// ---------------- grid barrier (agent scope, monotonic counter) ----------------
__device__ inline void grid_barrier(unsigned int* bar, unsigned int* phase) {
  __threadfence();               // make this block's stores device-visible
  __syncthreads();
  // Cluster user barrier: architecturally NOP when ClusterID==0 (normal dispatch).
  __builtin_amdgcn_s_cluster_barrier();
  if (threadIdx.x == 0) {
    *phase += NWG;
    __hip_atomic_fetch_add(bar, 1u, __ATOMIC_ACQ_REL, __HIP_MEMORY_SCOPE_AGENT);
    while (__hip_atomic_load(bar, __ATOMIC_ACQUIRE, __HIP_MEMORY_SCOPE_AGENT) < *phase)
      __builtin_amdgcn_s_sleep(1);
  }
  __syncthreads();
  __threadfence();               // acquire side: observe other blocks' stores
}

// A fragment from an LDS-resident 16x512 bf16 panel (row = lane%16)
__device__ inline v16bf lds_a_frag(const __bf16* panel_row, int kb, int hi) {
  v8bf lo = *(const v8bf*)(panel_row + kb + hi * 8);
  v8bf hf = *(const v8bf*)(panel_row + kb + 16 + hi * 8);
  v16bf a;
  #pragma unroll
  for (int i = 0; i < 8; ++i) { a[i] = lo[i]; a[8 + i] = hf[i]; }
  return a;
}

// ---------------- phase 2: persistent recurrence ----------------
// 64 WGs x 4 waves = 256 waves. The 4 waves of a WG share the same 16 M-rows, so the
// A panel (16x512 bf16 = 16KB) is staged in LDS once per phase and read via ds_load.
// Per step:
//  A) 256 tiles (1/wave): R = sigmoid(Xr + H@W_hr), Z = sigmoid(Xz + H@W_hz)
//  B) 128 tiles (blocks 0..31): Ht = tanh(Xh + (H*R)@W_hh); H = Z*H + (1-R)*Ht
__global__ __launch_bounds__(128) void gru_recur_kernel(
    const float* __restrict__ H0, const __bf16* __restrict__ wpk,
    const __bf16* __restrict__ xbuf, float* __restrict__ Hbuf,
    float* __restrict__ Rbuf, float* __restrict__ ZHbuf,
    __bf16* __restrict__ HRbuf, __bf16* __restrict__ Hbf,
    float* __restrict__ out, unsigned int* bar) {
  __shared__ __bf16 ldsA[16 * 512];          // 16KB A panel

  int tid  = threadIdx.x;
  int wid  = tid >> 5;
  int lane = tid & 31;
  int hi   = lane >> 4;
  int lm   = lane & 15;
  int gv   = blockIdx.x * 4 + wid;           // 0..255
  unsigned int phase = 0;

  // phase-A block-uniform tile geometry: gate/mt identical for the 4 waves of a WG
  int gateA = gv >> 7;                       // 0=R, 1=Z (uniform per block)
  int t128  = gv & 127;
  int mtA   = t128 >> 5;                     // uniform per block
  int ntA   = t128 & 31;
  // phase-B geometry (blocks 0..31 only)
  int mtB   = (gv < 128) ? (gv >> 5) : 0;
  int ntB   = gv & 31;

  // cooperative H0 -> Hbuf (f32) and Hbf (bf16 GEMM operand copy)
  for (int i = blockIdx.x * 128 + tid; i < Bn * Hn; i += NWG * 128) {
    float h = H0[i];
    Hbuf[i] = h;
    Hbf[i]  = (__bf16)h;
  }
  grid_barrier(bar, &phase);

  for (int t = 0; t < Tn; ++t) {
    size_t toff = (size_t)t * Bn * Hn;

    // ---- phase A: stage H panel to LDS, then 1 R/Z tile per wave ----
    {
      // copy Hbf rows [mtA*16, +16) into LDS (16KB, 128 lanes x 8 v8bf chunks)
      const v8bf* src = (const v8bf*)(Hbf + (size_t)mtA * 16 * Hn);
      v8bf* dst = (v8bf*)ldsA;
      #pragma unroll
      for (int c = 0; c < 8; ++c) dst[tid + c * 128] = src[tid + c * 128];
      __syncthreads();

      int n = ntA * 16 + lm;
      const __bf16* Xg = xbuf + (size_t)gateA * TBH + toff;   // Xr or Xz
      const __bf16* wb = wpk + (size_t)(3 + gateA) * WMAT + (size_t)ntA * 16 * 512;

      v8f acc;
      #pragma unroll
      for (int r = 0; r < 8; ++r)
        acc[r] = (float)Xg[(size_t)(mtA * 16 + r + 8 * hi) * Hn + n];

      const __bf16* prow = ldsA + lm * 512;
      v16bf bcur = *((const v16bf*)wb + lane);
      v16bf acur = lds_a_frag(prow, 0, hi);
      #pragma unroll
      for (int kt = 0; kt < 16; ++kt) {
        v16bf bnxt, anxt;
        if (kt < 15) {                       // 1-deep pipeline: prefetch kt+1
          bnxt = *((const v16bf*)(wb + (size_t)(kt + 1) * 512) + lane);
          anxt = lds_a_frag(prow, (kt + 1) * 32, hi);
        }
        acc = __builtin_amdgcn_wmma_f32_16x16x32_bf16(
            false, acur, false, bcur, (short)0, acc, false, false);
        acur = anxt; bcur = bnxt;
      }
      #pragma unroll
      for (int r = 0; r < 8; ++r) {
        int mm  = mtA * 16 + r + 8 * hi;
        float g = 1.0f / (1.0f + __expf(-acc[r]));
        float h = Hbuf[(size_t)mm * Hn + n];
        if (gateA == 0) {
          Rbuf[(size_t)mm * Hn + n]  = g;
          HRbuf[(size_t)mm * Hn + n] = (__bf16)(h * g);
        } else {
          ZHbuf[(size_t)mm * Hn + n] = g * h;
        }
      }
    }
    grid_barrier(bar, &phase);

    // ---- phase B: H_tilde tiles on blocks 0..31 ----
    if (blockIdx.x < 32) {
      const v8bf* src = (const v8bf*)(HRbuf + (size_t)mtB * 16 * Hn);
      v8bf* dst = (v8bf*)ldsA;
      #pragma unroll
      for (int c = 0; c < 8; ++c) dst[tid + c * 128] = src[tid + c * 128];
      __syncthreads();

      int n = ntB * 16 + lm;
      const __bf16* Xh = xbuf + 2 * TBH + toff;
      const __bf16* wb = wpk + 5 * WMAT + (size_t)ntB * 16 * 512;

      v8f acc;
      #pragma unroll
      for (int r = 0; r < 8; ++r)
        acc[r] = (float)Xh[(size_t)(mtB * 16 + r + 8 * hi) * Hn + n];

      const __bf16* prow = ldsA + lm * 512;
      v16bf bcur = *((const v16bf*)wb + lane);
      v16bf acur = lds_a_frag(prow, 0, hi);
      #pragma unroll
      for (int kt = 0; kt < 16; ++kt) {
        v16bf bnxt, anxt;
        if (kt < 15) {
          bnxt = *((const v16bf*)(wb + (size_t)(kt + 1) * 512) + lane);
          anxt = lds_a_frag(prow, (kt + 1) * 32, hi);
        }
        acc = __builtin_amdgcn_wmma_f32_16x16x32_bf16(
            false, acur, false, bcur, (short)0, acc, false, false);
        acur = anxt; bcur = bnxt;
      }
      #pragma unroll
      for (int r = 0; r < 8; ++r) {
        int mm    = mtB * 16 + r + 8 * hi;
        float ht  = tanhf(acc[r]);
        float rr  = Rbuf[(size_t)mm * Hn + n];
        float hn  = ZHbuf[(size_t)mm * Hn + n] + (1.0f - rr) * ht;  // Z*H + (1-R)*Ht
        Hbuf[(size_t)mm * Hn + n] = hn;
        Hbf[(size_t)mm * Hn + n]  = (__bf16)hn;
        out[toff + (size_t)mm * Hn + n] = hn;
        if (t == Tn - 1) out[(size_t)Tn * Bn * Hn + (size_t)mm * Hn + n] = hn;
      }
    }
    grid_barrier(bar, &phase);
  }
}

// ---------------- host launcher ----------------
extern "C" void kernel_launch(void* const* d_in, const int* in_sizes, int n_in,
                              void* d_out, int out_size, void* d_ws, size_t ws_size,
                              hipStream_t stream) {
  (void)in_sizes; (void)n_in; (void)out_size; (void)ws_size;
  const float* inp  = (const float*)d_in[0];
  const float* H0   = (const float*)d_in[1];
  const float* W_xr = (const float*)d_in[2];
  const float* W_hr = (const float*)d_in[3];
  const float* b_r  = (const float*)d_in[4];
  const float* W_xz = (const float*)d_in[5];
  const float* W_hz = (const float*)d_in[6];
  const float* b_z  = (const float*)d_in[7];
  const float* W_xh = (const float*)d_in[8];
  const float* W_hh = (const float*)d_in[9];
  const float* b_h  = (const float*)d_in[10];

  char* ws = (char*)d_ws;
  unsigned int* bar = (unsigned int*)(ws + BAR_OFF);
  __bf16* wpk   = (__bf16*)(ws + W_OFF);
  __bf16* xbuf  = (__bf16*)(ws + X_OFF);
  float*  Hbuf  = (float*)(ws + H_OFF);
  float*  Rbuf  = (float*)(ws + R_OFF);
  float*  ZHbuf = (float*)(ws + ZH_OFF);
  __bf16* HRbuf = (__bf16*)(ws + HR_OFF);
  __bf16* Hbf   = (__bf16*)(ws + HBF_OFF);

  hipMemsetAsync(bar, 0, 256, stream);  // reset barrier counter (graph-capture safe)

  // pack order: W_xr, W_xz, W_xh, W_hr, W_hz, W_hh
  pack_w_kernel<<<(int)((6 * WMAT + 255) / 256), 256, 0, stream>>>(
      W_xr, W_xz, W_xh, W_hr, W_hz, W_hh, wpk);

  // 4096 M-tiles * 8 N-groups = 32768 waves / 8 per block; 3 gates fused per wave
  xgemm_kernel<<<4096, 256, 0, stream>>>(inp, wpk, b_r, b_z, b_h, xbuf);

  gru_recur_kernel<<<NWG, 128, 0, stream>>>(
      H0, wpk, xbuf, Hbuf, Rbuf, ZHbuf, HRbuf, Hbf, (float*)d_out, bar);
}
